// QuantLinear_65901978190254
// MI455X (gfx1250) — compile-verified, matrix operand order checked by reading, not run on hardware
//
#include <hip/hip_runtime.h>
#include <stdint.h>

#define IN_F      4096
#define OUT_F     11008
#define BM        128
#define BN        128
#define BK        32
#define LDA       40            // padded LDS row stride (elements); 80B keeps 16B alignment, skews banks
#define KSTEPS    (IN_F / BK)   // 128

#if __has_builtin(__builtin_amdgcn_global_load_async_to_lds_b128)
#define HAVE_ASYNC_LDS 1
#else
#define HAVE_ASYNC_LDS 0
#endif

#define AS_GLOBAL __attribute__((address_space(1)))
#define AS_LOCAL  __attribute__((address_space(3)))

typedef __bf16 bf16x16 __attribute__((ext_vector_type(16)));
typedef __bf16 bf16x2  __attribute__((ext_vector_type(2)));
typedef float  v8f     __attribute__((ext_vector_type(8)));
typedef int    v4i     __attribute__((ext_vector_type(4)));

union Frag  { bf16x16 v; uint4 u[2]; };
union Pack8 { bf16x2 h[4]; uint4 q; };

__device__ __forceinline__ void wait_async0() {
#if HAVE_ASYNC_LDS
#if __has_builtin(__builtin_amdgcn_s_wait_asynccnt)
    __builtin_amdgcn_s_wait_asynccnt(0);
#else
    asm volatile("s_wait_asynccnt 0x0" ::: "memory");
#endif
#endif
}

// packed fp32 -> bf16 pair, via HW packed convert when available
__device__ __forceinline__ bf16x2 pkbf(float a, float b) {
#if __has_builtin(__builtin_amdgcn_cvt_pk_bf16_f32)
    return __builtin_amdgcn_cvt_pk_bf16_f32(a, b);
#else
    bf16x2 r; r[0] = (__bf16)a; r[1] = (__bf16)b; return r;
#endif
}

// 8 fp32 -> 8 packed bf16
__device__ __forceinline__ uint4 cvt8(float4 a, float4 b) {
    Pack8 p;
    p.h[0] = pkbf(a.x, a.y); p.h[1] = pkbf(a.z, a.w);
    p.h[2] = pkbf(b.x, b.y); p.h[3] = pkbf(b.z, b.w);
    return p.q;
}

// unpack one packed int32 (8 nibbles along K) -> 8 bf16 of fma(q, s, -s*z).
__device__ __forceinline__ uint4 dq8(unsigned w, float s, float nsz) {
    unsigned lo = w & 0x0F0F0F0Fu;
    unsigned hi = (w >> 4u) & 0x0F0F0F0Fu;
    float f0 = fmaf((float)( lo          & 0xFFu), s, nsz);
    float f1 = fmaf((float)( hi          & 0xFFu), s, nsz);
    float f2 = fmaf((float)((lo >>  8u) & 0xFFu), s, nsz);
    float f3 = fmaf((float)((hi >>  8u) & 0xFFu), s, nsz);
    float f4 = fmaf((float)((lo >> 16u) & 0xFFu), s, nsz);
    float f5 = fmaf((float)((hi >> 16u) & 0xFFu), s, nsz);
    float f6 = fmaf((float)( lo >> 24u        ), s, nsz);
    float f7 = fmaf((float)( hi >> 24u        ), s, nsz);
    Pack8 p;
    p.h[0] = pkbf(f0, f1); p.h[1] = pkbf(f2, f3);
    p.h[2] = pkbf(f4, f5); p.h[3] = pkbf(f6, f7);
    return p.q;
}

// issue a 16B global->LDS async copy (or sync fallback)
__device__ __forceinline__ void copyA16(const unsigned short* g, unsigned short* l) {
#if HAVE_ASYNC_LDS
    __builtin_amdgcn_global_load_async_to_lds_b128(
        (AS_GLOBAL v4i*)g, (AS_LOCAL v4i*)l, 0, 0);
#else
    *(uint4*)l = *(const uint4*)g;
#endif
}

// ---------------- prepass: x fp32 -> bf16 (into workspace) ----------------
__global__ __launch_bounds__(256)
void x_to_bf16(const float* __restrict__ x, unsigned short* __restrict__ xb) {
    const size_t i = (size_t)blockIdx.x * blockDim.x + threadIdx.x;   // one thread = 8 elements
    const float4* p = (const float4*)x + i * 2;
    float4 a = p[0], b = p[1];
    *(uint4*)(xb + i * 8) = cvt8(a, b);
}

// ---------------- fused GPTQ dequant + bf16 WMMA GEMM ----------------
// ABF=true : A source is pre-converted bf16 (async direct-to-LDS staging)
// ABF=false: A source is fp32, converted on the fly while staging to LDS
template <bool ABF>
__global__ __launch_bounds__(256)
void gptq_wmma_gemm(const void*  __restrict__ xin,
                    const int*   __restrict__ qweight,
                    const int*   __restrict__ qzeros,
                    const float* __restrict__ scales,
                    const float* __restrict__ bias,
                    float*       __restrict__ out)
{
    __shared__ unsigned short sA[2][BM * LDA];   // A tile, [m][k] bf16
    __shared__ unsigned short sB[2][BN * LDA];   // B tile, [n][k] bf16 (K contiguous per n)

    const int tid   = threadIdx.x;
    const int lane  = tid & 31;
    const int wid   = tid >> 5;       // 0..7
    const int waveM = wid >> 1;       // 0..3 -> 32 rows each
    const int waveN = wid & 1;        // 0..1 -> 64 cols each

    const int m0 = blockIdx.y * BM;
    const int n0 = blockIdx.x * BN;

    // A staging map: thread -> (row, 16 contiguous k)
    const int arow = tid >> 1;                 // 0..127
    const int acol = (tid & 1) << 4;           // 0 or 16
    const float*          aptrF = (const float*)xin          + (size_t)(m0 + arow) * IN_F + acol;
    const unsigned short* aptrH = (const unsigned short*)xin + (size_t)(m0 + arow) * IN_F + acol;

    // B staging map: thread -> fixed output col, two packed int32 rows (16 K values)
    const int nl     = tid & 127;              // local col 0..127
    const int qsel   = tid >> 7;               // 0..1 -> K offset 0 or 16
    const int ncol   = n0 + nl;
    const int zshift = (ncol & 7) * 4;

    v8f acc[2][4];
    const v8f vzero = {0.f,0.f,0.f,0.f,0.f,0.f,0.f,0.f};
#pragma unroll
    for (int i = 0; i < 2; ++i)
#pragma unroll
        for (int j = 0; j < 4; ++j) acc[i][j] = vzero;

    // group-persistent dequant constants (group = k/128 = ks/4)
    float sc  = scales[ncol];
    float nsz = -sc * (float)((qzeros[ncol >> 3] >> zshift) & 15);

    // ---------------- prologue: stage K-step 0 into buffer 0 ----------------
    {
        unsigned short* da = &sA[0][arow * LDA + acol];
        if constexpr (ABF) {
            copyA16(aptrH,     da);
            copyA16(aptrH + 8, da + 8);
        } else {
            const float4* p = (const float4*)aptrF;
            *(uint4*)(da)     = cvt8(p[0], p[1]);
            *(uint4*)(da + 8) = cvt8(p[2], p[3]);
        }
        const int qr = qsel * 2;
        unsigned w0 = (unsigned)qweight[(size_t)qr * OUT_F + ncol];
        unsigned w1 = (unsigned)qweight[(size_t)(qr + 1) * OUT_F + ncol];
        unsigned short* db = &sB[0][nl * LDA + qsel * 16];
        *(uint4*)(db)     = dq8(w0, sc, nsz);
        *(uint4*)(db + 8) = dq8(w1, sc, nsz);
    }
    if constexpr (ABF) wait_async0();
    __syncthreads();

    const int lo   = (lane < 16) ? 0 : 1;   // half-wave select
    const int lrow = lane & 15;

    // ---------------- main loop, LDS double buffered ----------------
    for (int ks = 0; ks < KSTEPS; ++ks) {
        const int cur = ks & 1, nxt = cur ^ 1;
        const bool has = (ks + 1) < KSTEPS;

        // prefetch next tile while WMMAs run.
        // (A in ABF mode: async direct-to-LDS, safe because the barrier that
        //  ended iteration ks-1 guarantees no wave still reads buffer `nxt`.)
        float4 t0 = {0,0,0,0}, t1 = t0, t2 = t0, t3 = t0;
        unsigned w0 = 0, w1 = 0;
        if (has) {
            if constexpr (ABF) {
                const unsigned short* g = aptrH + (size_t)(ks + 1) * BK;
                unsigned short* da = &sA[nxt][arow * LDA + acol];
                copyA16(g,     da);
                copyA16(g + 8, da + 8);
            } else {
                const float4* p = (const float4*)(aptrF + (size_t)(ks + 1) * BK);
                t0 = p[0]; t1 = p[1]; t2 = p[2]; t3 = p[3];
            }
            const int qr = (ks + 1) * 4 + qsel * 2;
            w0 = (unsigned)qweight[(size_t)qr * OUT_F + ncol];
            w1 = (unsigned)qweight[(size_t)(qr + 1) * OUT_F + ncol];
            if (((ks + 1) & 3) == 0) {                          // new quant group
                const int g2 = (ks + 1) >> 2;
                sc  = scales[(size_t)g2 * OUT_F + ncol];
                nsz = -sc * (float)((qzeros[(size_t)g2 * (OUT_F / 8) + (ncol >> 3)] >> zshift) & 15);
            }
        }

        // fragment loads: two ds_load_b128 per 16x16 fragment
        Frag aF[2], bF[4];
#pragma unroll
        for (int i = 0; i < 2; ++i) {
            const unsigned short* base = &sA[cur][(waveM * 32 + i * 16 + lrow) * LDA];
            aF[i].u[0] = *(const uint4*)(base + (lo ? 8 : 0));
            aF[i].u[1] = *(const uint4*)(base + (lo ? 24 : 16));
        }
#pragma unroll
        for (int j = 0; j < 4; ++j) {
            const unsigned short* base = &sB[cur][(waveN * 64 + j * 16 + lrow) * LDA];
            bF[j].u[0] = *(const uint4*)(base + (lo ? 8 : 0));
            bF[j].u[1] = *(const uint4*)(base + (lo ? 24 : 16));
        }

#pragma unroll
        for (int i = 0; i < 2; ++i)
#pragma unroll
            for (int j = 0; j < 4; ++j)
                acc[i][j] = __builtin_amdgcn_wmma_f32_16x16x32_bf16(
                    false, aF[i].v, false, bF[j].v, (short)0, acc[i][j], false, false);

        if (has) {
            if constexpr (!ABF) {
                unsigned short* da = &sA[nxt][arow * LDA + acol];
                *(uint4*)(da)     = cvt8(t0, t1);
                *(uint4*)(da + 8) = cvt8(t2, t3);
            }
            unsigned short* db = &sB[nxt][nl * LDA + qsel * 16];
            *(uint4*)(db)     = dq8(w0, sc, nsz);
            *(uint4*)(db + 8) = dq8(w1, sc, nsz);
        }
        if constexpr (ABF) wait_async0();
        __syncthreads();
    }

    // ---------------- epilogue: bias + store (C layout: VGPR r -> M = r + 8*(lane>=16)) ----------------
    const int rbase = lo * 8;
#pragma unroll
    for (int j = 0; j < 4; ++j) {
        const int col = n0 + waveN * 64 + j * 16 + lrow;
        const float bv = bias[col];
#pragma unroll
        for (int i = 0; i < 2; ++i) {
            const int row0 = m0 + waveM * 32 + i * 16 + rbase;
            float* o = out + (size_t)row0 * OUT_F + col;
#pragma unroll
            for (int r = 0; r < 8; ++r)
                o[(size_t)r * OUT_F] = acc[i][j][r] + bv;
        }
    }
}

extern "C" void kernel_launch(void* const* d_in, const int* in_sizes, int n_in,
                              void* d_out, int out_size, void* d_ws, size_t ws_size,
                              hipStream_t stream) {
    const float* x       = (const float*)d_in[0];
    const int*   qweight = (const int*)d_in[1];
    const int*   qzeros  = (const int*)d_in[2];
    const float* scales  = (const float*)d_in[3];
    const float* bias    = (const float*)d_in[4];
    float*       out     = (float*)d_out;

    const int rowsM = in_sizes[0] / IN_F;               // 4*2048 = 8192
    dim3 grid(OUT_F / BN, rowsM / BM);                  // (86, 64)

    const size_t elemsA = (size_t)rowsM * IN_F;         // 33.5M
    const size_t needA  = elemsA * sizeof(unsigned short);

    if (ws_size >= needA) {
        // one-shot fp32 -> bf16 conversion of x, then bf16-A GEMM
        unsigned short* xb = (unsigned short*)d_ws;
        const int nThreads8 = (int)(elemsA / 8);        // 8 elements per thread
        x_to_bf16<<<nThreads8 / 256, 256, 0, stream>>>(x, xb);
        gptq_wmma_gemm<true><<<grid, 256, 0, stream>>>(xb, qweight, qzeros, scales, bias, out);
    } else {
        gptq_wmma_gemm<false><<<grid, 256, 0, stream>>>(x, qweight, qzeros, scales, bias, out);
    }
}
